// QWKLoss_78443282694922
// MI455X (gfx1250) — compile-verified, matrix operand order checked by reading, not run on hardware
//
#include <hip/hip_runtime.h>
#include <stdint.h>

typedef __attribute__((ext_vector_type(8))) int   v8i;
typedef __attribute__((ext_vector_type(4))) float v4f;
typedef __attribute__((ext_vector_type(4))) int   v4i;

// Byte-equality mask: 0x80 in each byte of w equal to the byte replicated in
// qrep, 0x00 elsewhere. Exact; lowers to xor + bitop3 + add + bitop3.
__device__ __forceinline__ uint32_t eq4b(uint32_t w, uint32_t qrep) {
    uint32_t x = w ^ qrep;                               // 0x00 where equal
    uint32_t t = (x & 0x7F7F7F7Fu) + 0x7F7F7F7Fu;        // bit7 <- (low7 != 0)
    return ~(t | x) & 0x80808080u;                       // 0x80 at equal bytes
}

__device__ __forceinline__ unsigned argmax5(float l0, float l1, float l2,
                                            float l3, float l4) {
    int pi = 0; float best = l0;                         // first-max ties (jnp)
    if (l1 > best) { best = l1; pi = 1; }
    if (l2 > best) { best = l2; pi = 2; }
    if (l3 > best) { best = l3; pi = 3; }
    if (l4 > best) { best = l4; pi = 4; }
    return (unsigned)pi;
}

// Pack 4 consecutive samples' cell indices c = t*5 + p into one word.
// OOB samples get 0xFF (matches no row).
__device__ __forceinline__ uint32_t pack_cw(const float* __restrict__ logits,
                                            const int*   __restrict__ targets,
                                            int s0, int n) {
    uint32_t cw = 0xFFFFFFFFu;
    if (s0 + 3 < n) {                                    // fast path: aligned b128
        const v4f* lp = (const v4f*)(logits + (size_t)s0 * 5u);
        v4f q0 = __builtin_nontemporal_load(lp + 0);
        v4f q1 = __builtin_nontemporal_load(lp + 1);
        v4f q2 = __builtin_nontemporal_load(lp + 2);
        v4f q3 = __builtin_nontemporal_load(lp + 3);
        v4f q4 = __builtin_nontemporal_load(lp + 4);
        v4i tv = __builtin_nontemporal_load((const v4i*)(targets + s0));
        unsigned c0 = (unsigned)tv[0] * 5u + argmax5(q0[0], q0[1], q0[2], q0[3], q1[0]);
        unsigned c1 = (unsigned)tv[1] * 5u + argmax5(q1[1], q1[2], q1[3], q2[0], q2[1]);
        unsigned c2 = (unsigned)tv[2] * 5u + argmax5(q2[2], q2[3], q3[0], q3[1], q3[2]);
        unsigned c3 = (unsigned)tv[3] * 5u + argmax5(q3[3], q4[0], q4[1], q4[2], q4[3]);
        cw = c0 | (c1 << 8) | (c2 << 16) | (c3 << 24);
    } else if (s0 < n) {                                 // rare ragged tail
        cw = 0;
        #pragma unroll
        for (int i = 0; i < 4; ++i) {
            unsigned c = 0xFFu;
            const int s = s0 + i;
            if (s < n) {
                const float* lp = logits + (size_t)s * 5u;
                c = (unsigned)targets[s] * 5u +
                    argmax5(lp[0], lp[1], lp[2], lp[3], lp[4]);
            }
            cw |= c << (8 * i);
        }
    }
    return cw;
}

__global__ void qwk_zero_ws(int* __restrict__ ws) {
    if (threadIdx.x < 32) ws[threadIdx.x] = 0;
}

// 25-cell histogram of c = target*5 + pred via V_WMMA_I32_16X16X64_IU8.
// A = one-hot(c) (rows 0-15 / 16-31 -> acc1/acc2), B = all-ones constant.
// One-hot bytes are 0x80, so accumulated counts are scaled by 128 (exact).
// 256 samples per wave-iteration: two independent dataflow streams so the
// scheduler can fill WMMA->VALU hazard slots and DS/VMEM latency.
__global__ __launch_bounds__(256) void qwk_confusion(
        const float* __restrict__ logits,
        const int*   __restrict__ targets,
        int*         __restrict__ ws,
        int n) {
    __shared__ int sH[8 * 32];

    const int  tid  = threadIdx.x;
    const int  lane = tid & 31;
    const int  w    = tid >> 5;
    const int  hib  = (lane >= 16) ? 8 : 0;              // A-layout hi-half byte off
    const uint32_t qrep  = (uint32_t)(lane & 15) * 0x01010101u;  // rows 0..15
    const uint32_t qrep2 = qrep + 0x10101010u;                   // rows 16..31

    v8i bones;                                           // B[k][n] = 1 for all k,n
    #pragma unroll
    for (int j = 0; j < 8; ++j) bones[j] = 0x01010101;

    v8i acc1 = {0,0,0,0,0,0,0,0};
    v8i acc2 = {0,0,0,0,0,0,0,0};

    const int nChunk = (n + 255) >> 8;                   // 256 samples / wave-chunk
    const int gw     = blockIdx.x * 8 + w;
    const int gstep  = gridDim.x * 8;

    for (int chunk = gw; chunk < nChunk; chunk += gstep) {
        const int base = chunk << 8;

        // ---- two packed words: samples [base, base+128) and [base+128, base+256)
        uint32_t cwA = pack_cw(logits, targets, base + (lane << 2), n);
        uint32_t cwB = pack_cw(logits, targets, base + 128 + (lane << 2), n);

        // ---- four K=64 phases; A dword (j, half) comes from packed word of
        // lane g = (q&1)*16 + (j>>1)*4 + (j&1) + (hi?2:0)  ->  one ds_bpermute.
        #pragma unroll
        for (int q = 0; q < 4; ++q) {
            const int src = (int)((q < 2) ? cwA : cwB);
            v8i a1, a2;
            #pragma unroll
            for (int j = 0; j < 8; ++j) {
                const int gb = hib + 4 * ((q & 1) * 16 + ((j >> 1) << 2) + (j & 1));
                const uint32_t bw =
                    (uint32_t)__builtin_amdgcn_ds_bpermute(gb, src);
                a1[j] = (int)eq4b(bw, qrep);             // rows 0..15
                a2[j] = (int)eq4b(bw, qrep2);            // rows 16..31
            }
            acc1 = __builtin_amdgcn_wmma_i32_16x16x64_iu8(false, a1, false, bones,
                                                          acc1, false, false);
            acc2 = __builtin_amdgcn_wmma_i32_16x16x64_iu8(false, a2, false, bones,
                                                          acc2, false, false);
        }
    }

    // D[m][n] identical for all n (B all-ones); read column 0:
    // lane 0 VGPR r -> m=r, lane 16 VGPR r -> m=8+r.
    if (lane == 0 || lane == 16) {
        const int off = w * 32 + ((lane == 16) ? 8 : 0);
        #pragma unroll
        for (int r = 0; r < 8; ++r) {
            sH[off + r]      = acc1[r];                  // c = m
            sH[off + 16 + r] = acc2[r];                  // c = 16 + m
        }
    }
    __syncthreads();
    if (tid < 25) {
        int sum = 0;
        #pragma unroll
        for (int ww = 0; ww < 8; ++ww) sum += sH[ww * 32 + tid];
        atomicAdd(&ws[tid], sum);                        // values scaled by 128
    }
}

__global__ void qwk_finalize(const int* __restrict__ ws, float* __restrict__ out) {
    if (threadIdx.x == 0 && blockIdx.x == 0) {
        double H[5][5];
        double r[5] = {0, 0, 0, 0, 0}, c[5] = {0, 0, 0, 0, 0}, T = 0.0;
        for (int m = 0; m < 5; ++m)
            for (int nn = 0; nn < 5; ++nn) {
                double h = (double)ws[m * 5 + nn] * (1.0 / 128.0); // un-scale
                H[m][nn] = h; r[m] += h; c[nn] += h; T += h;
            }
        double num = 0.0, den = 0.0;
        for (int m = 0; m < 5; ++m)
            for (int nn = 0; nn < 5; ++nn) {
                double wt = (double)((m - nn) * (m - nn)) * (1.0 / 16.0);
                num += wt * H[m][nn];
                den += wt * r[m] * c[nn];
            }
        den = (T > 0.0) ? (den / T) : 0.0;
        out[0] = (float)(num / (den + 1e-6));            // == 1 - qwk
    }
}

extern "C" void kernel_launch(void* const* d_in, const int* in_sizes, int n_in,
                              void* d_out, int out_size, void* d_ws, size_t ws_size,
                              hipStream_t stream) {
    const float* logits  = (const float*)d_in[0];
    const int*   targets = (const int*)d_in[1];
    const int    n       = in_sizes[1];                  // N samples
    int*   ws  = (int*)d_ws;
    float* out = (float*)d_out;

    qwk_zero_ws<<<1, 32, 0, stream>>>(ws);

    const int nChunk = (n + 255) >> 8;                   // 256 samples per wave
    int blocks = (nChunk + 7) / 8;
    if (blocks > 2048) blocks = 2048;
    if (blocks < 1) blocks = 1;
    qwk_confusion<<<blocks, 256, 0, stream>>>(logits, targets, ws, n);

    qwk_finalize<<<1, 1, 0, stream>>>(ws, out);
}
